// MultiHeadAttention_Slow_6339371729115
// MI455X (gfx1250) — matrix operand order repaired and checked by reading, not compiled, not run on hardware
//
#include <hip/hip_runtime.h>
#include <stdint.h>

typedef _Float16 half_t;
typedef __attribute__((ext_vector_type(16))) _Float16 v16h;
typedef __attribute__((ext_vector_type(8)))  _Float16 v8h;
typedef __attribute__((ext_vector_type(8)))  float    v8f;
typedef __attribute__((ext_vector_type(4)))  unsigned int u32x4;
typedef __attribute__((ext_vector_type(4)))  int          i32x4;
typedef __attribute__((ext_vector_type(8)))  int          i32x8;

namespace {
constexpr int S_LEN = 2048;
constexpr int EMB   = 2048;
constexpr int NH    = 16;
constexpr int DHEAD = 128;
constexpr float SOFT_SCALE = 0.08838834764831845f; // 1/sqrt(128)
}

// ---------------------------------------------------------------------------
// Tensor Data Mover support (CDNA5). Gated on builtin availability; arity
// differs between ROCm 7.2 (5 args) and amdgpu-toolchain (6 args) -- the
// 6-arg headers ship the gfx1250 TDM header, so use that as the discriminator.
// ---------------------------------------------------------------------------
#if defined(__has_builtin)
#  if __has_builtin(__builtin_amdgcn_tensor_load_to_lds)
#    define HAVE_TDM 1
#  endif
#endif
#ifndef HAVE_TDM
#  define HAVE_TDM 0
#endif
#if __has_include(<hip/amd_detail/amd_gfx1250_TDM.h>)
#  define TDM_6ARGS 1
#else
#  define TDM_6ARGS 0
#endif

#if HAVE_TDM
// 2D f16 tile load: tile_d0 (contiguous elements) x tile_d1 rows, row stride
// stride0 (elements) in global memory; LDS destination packed row-major with
// optional padding (pad codes per D# group1 encoding).
__device__ __forceinline__ void tdm_load_2d_f16(uint32_t lds_addr, const void* gptr,
                                                uint32_t tensor_d0, uint32_t tensor_d1,
                                                uint32_t stride0,
                                                uint32_t tile_d0, uint32_t tile_d1,
                                                uint32_t pad_en, uint32_t pad_int_code,
                                                uint32_t pad_amt_code)
{
  const uint64_t ga = (uint64_t)(uintptr_t)gptr;
  u32x4 g0;
  g0[0] = 1u;                                    // count=1, is_restore=0, no gather
  g0[1] = lds_addr;                              // bits 63:32
  g0[2] = (uint32_t)(ga & 0xFFFFFFFFu);          // global_addr[31:0]
  g0[3] = (uint32_t)((ga >> 32) & 0x01FFFFFFu)   // global_addr[56:32]
        | (2u << 30);                            // type = 2 ("image")
  i32x8 g1;
  g1[0] = (int)((1u << 16)                       // data_size = 2 bytes
              | (pad_en << 20)
              | (pad_int_code << 22)
              | (pad_amt_code << 25));           // workgroup_mask = 0
  g1[1] = (int)((tensor_d0 & 0xFFFFu) << 16);    // tensor_dim0[15:0] @ bits 63:48
  g1[2] = (int)(((tensor_d0 >> 16) & 0xFFFFu)    // tensor_dim0[31:16]
              | ((tensor_d1 & 0xFFFFu) << 16));  // tensor_dim1[15:0]
  g1[3] = (int)(((tensor_d1 >> 16) & 0xFFFFu)    // tensor_dim1[31:16]
              | ((tile_d0 & 0xFFFFu) << 16));    // tile_dim0
  g1[4] = (int)(tile_d1 & 0xFFFFu);              // tile_dim1 ; tile_dim2 = 0
  g1[5] = (int)stride0;                          // tensor_dim0_stride[31:0]
  g1[6] = 0;                                     // stride0[47:32] | stride1[15:0]
  g1[7] = 0;                                     // stride1[47:16]
  i32x4 z4; z4[0] = z4[1] = z4[2] = z4[3] = 0;
#if TDM_6ARGS
  i32x8 z8; z8[0]=z8[1]=z8[2]=z8[3]=z8[4]=z8[5]=z8[6]=z8[7]=0;
  __builtin_amdgcn_tensor_load_to_lds(g0, g1, z4, z4, z8, 0);
#else
  __builtin_amdgcn_tensor_load_to_lds(g0, g1, z4, z4, 0);
#endif
}
#endif // HAVE_TDM

// ---------------------------------------------------------------------------
// WMMA fragment loaders (wave32, V_WMMA_F32_16X16X32_F16)
// ---------------------------------------------------------------------------
__device__ inline v16h frag_a(const half_t* base, int ldk) {
  const int lane = threadIdx.x & 31;
  const int m = lane & 15;
  const int g = lane >> 4;
  const half_t* p = base + m * ldk + g * 8;
  v8h lo = *(const v8h*)(p);
  v8h hi = *(const v8h*)(p + 16);
  v16h a;
#pragma unroll
  for (int i = 0; i < 8; ++i) { a[i] = lo[i]; a[i + 8] = hi[i]; }
  return a;
}

__device__ inline v16h frag_b(const half_t* base, int ldk) {
  const int lane = threadIdx.x & 31;
  const int n = lane & 15;
  const int g = lane >> 4;
  const half_t* p = base + n * ldk + g * 8;
  v8h lo = *(const v8h*)(p);
  v8h hi = *(const v8h*)(p + 16);
  v16h b;
#pragma unroll
  for (int i = 0; i < 8; ++i) { b[i] = lo[i]; b[i + 8] = hi[i]; }
  return b;
}

// ---------------------------------------------------------------------------
// Elementwise f32 -> f16 cast
// ---------------------------------------------------------------------------
__global__ void __launch_bounds__(256)
cast_f32_to_f16(const float* __restrict__ src, half_t* __restrict__ dst, int n) {
  int i = blockIdx.x * blockDim.x + threadIdx.x;
  const int stride = gridDim.x * blockDim.x;
  for (; i < n; i += stride) dst[i] = (half_t)src[i];
}

// ---------------------------------------------------------------------------
// Generic tiled WMMA GEMM: C[M,N] = A[M,K] * B[K,N] + bias[N]
// Tile 128x128x32; 8 waves (4x2), wave tile 32x64. A tile staged via TDM.
// ---------------------------------------------------------------------------
__global__ void __launch_bounds__(256)
gemm_f16_wmma(const half_t* __restrict__ A, int lda,
              const half_t* __restrict__ B, int ldb, long strideB,
              const float* __restrict__ bias, long strideBias,
              half_t* __restrict__ outH, float* __restrict__ outF,
              int ldo, long strideOut, int Kdim)
{
  constexpr int BM = 128, BN = 128, BK = 32, LDK = BK + 8; // 80B rows, 16B aligned
  __shared__ __align__(16) half_t sA[BM * LDK];
  __shared__ __align__(16) half_t sB[BN * LDK]; // stored [N][K]

  const int tid  = threadIdx.x;
  const int lane = tid & 31;
  const int wave = tid >> 5;
  const int wm   = wave >> 1;   // 0..3 : 32-row group
  const int wn   = wave & 1;    // 0..1 : 64-col group
  const int cn   = lane & 15;
  const int g    = lane >> 4;

  const int row0 = blockIdx.x * BM;
  const int col0 = blockIdx.y * BN;
  const half_t* Bz = B + (long)blockIdx.z * strideB;
#if HAVE_TDM
  const uint32_t ldsA = (uint32_t)(uintptr_t)(void*)sA; // LDS aperture: addr[31:0]
#endif

  v8f acc[2][4];
#pragma unroll
  for (int i = 0; i < 2; ++i)
#pragma unroll
    for (int j = 0; j < 4; ++j)
#pragma unroll
      for (int v = 0; v < 8; ++v) acc[i][j][v] = 0.f;

  for (int k0 = 0; k0 < Kdim; k0 += BK) {
#if HAVE_TDM
    // A tile via Tensor Data Mover: 128 rows x 32 halfs, row stride lda;
    // pad 16 DWORDs (row) -> +4 DWORDs to realize LDK = 40 halfs.
    if (wave == 0)
      tdm_load_2d_f16(ldsA, A + (long)row0 * lda + k0,
                      /*tensor_d0=*/BK, /*tensor_d1=*/BM, /*stride0=*/(uint32_t)lda,
                      /*tile_d0=*/BK, /*tile_d1=*/BM,
                      /*pad_en=*/1u, /*pad_int=*/3u /*16 DW*/, /*pad_amt=*/3u /*4 DW*/);
#else
    { // stage A tile manually: 128 x 32 row-major; 16 halfs per thread
      const int r = tid >> 1;
      const int p = (tid & 1) * 16;
      const half_t* gp = A + (long)(row0 + r) * lda + k0 + p;
      *(v8h*)(sA + r * LDK + p)     = *(const v8h*)(gp);
      *(v8h*)(sA + r * LDK + p + 8) = *(const v8h*)(gp + 8);
    }
#endif
    { // stage B tile transposed: read B[k][n] coalesced, write sB[n][k]
      const int kk = tid >> 3;          // 0..31
      const int p  = (tid & 7) * 16;    // 0..112
      const half_t* gp = Bz + (long)(k0 + kk) * ldb + col0 + p;
      v8h x0 = *(const v8h*)(gp);
      v8h x1 = *(const v8h*)(gp + 8);
#pragma unroll
      for (int i = 0; i < 8; ++i) sB[(p + i)     * LDK + kk] = x0[i];
#pragma unroll
      for (int i = 0; i < 8; ++i) sB[(p + 8 + i) * LDK + kk] = x1[i];
    }
#if HAVE_TDM
    if (wave == 0) __builtin_amdgcn_s_wait_tensorcnt(0);
#endif
    __syncthreads();

    v16h af0 = frag_a(sA + (wm * 32)      * LDK, LDK);
    v16h af1 = frag_a(sA + (wm * 32 + 16) * LDK, LDK);
    v16h bf[4];
#pragma unroll
    for (int j = 0; j < 4; ++j) bf[j] = frag_b(sB + (wn * 64 + j * 16) * LDK, LDK);
#pragma unroll
    for (int j = 0; j < 4; ++j) {
      acc[0][j] = __builtin_amdgcn_wmma_f32_16x16x32_f16(false, af0, false, bf[j],
                                                         (short)0, acc[0][j], false, false);
      acc[1][j] = __builtin_amdgcn_wmma_f32_16x16x32_f16(false, af1, false, bf[j],
                                                         (short)0, acc[1][j], false, false);
    }
    __syncthreads();
  }

  const float* bz = bias + (long)blockIdx.z * strideBias;
  const long obase = (long)blockIdx.z * strideOut;
#pragma unroll
  for (int i = 0; i < 2; ++i) {
#pragma unroll
    for (int j = 0; j < 4; ++j) {
      const int col = col0 + wn * 64 + j * 16 + cn;
      const float bval = bz[col];
#pragma unroll
      for (int v = 0; v < 8; ++v) {
        const int r = row0 + wm * 32 + i * 16 + g * 8 + v;
        const float val = acc[i][j][v] + bval;
        if (outH) outH[obase + (long)r * ldo + col] = (half_t)val;
        else      outF[obase + (long)r * ldo + col] = val;
      }
    }
  }
}

// ---------------------------------------------------------------------------
// Causal flash attention, f16 WMMA, f32 accumulation, K tile via TDM.
// ---------------------------------------------------------------------------
__global__ void __launch_bounds__(256)
flash_attn_f16(const half_t* __restrict__ Q, const half_t* __restrict__ K,
               const half_t* __restrict__ V, half_t* __restrict__ attnOut)
{
  constexpr int QB = 128, KBLK = 64;
  constexpr int LKD = DHEAD + 8;  // 136 halfs (272B rows)
  constexpr int LVD = KBLK + 8;   // 72 halfs (144B rows)
  constexpr int LPD = KBLK + 8;   // 72
  __shared__ __align__(16) half_t sK [KBLK * LKD];      // 17408 B
  __shared__ __align__(16) half_t sVt[DHEAD * LVD];     // 18432 B
  __shared__ __align__(16) half_t sP [8 * 16 * LPD];    // 18432 B

  const int tid  = threadIdx.x;
  const int lane = tid & 31;
  const int wave = tid >> 5;
  const int cn   = lane & 15;
  const int g    = lane >> 4;

  const int h  = blockIdx.y;
  const int qb = blockIdx.x;
  const long headOff = (long)h * S_LEN * DHEAD;
  const half_t* Qh = Q + headOff;
  const half_t* Kh = K + headOff;
  const half_t* Vh = V + headOff;
#if HAVE_TDM
  const uint32_t ldsK = (uint32_t)(uintptr_t)(void*)sK;
#endif

  const int qrow0 = qb * QB + wave * 16;

  v16h qf[4]; // Q rows qrow0..+15, DH=128 as 4 K-chunks, straight from global
#pragma unroll
  for (int kc = 0; kc < 4; ++kc)
    qf[kc] = frag_a(Qh + (long)qrow0 * DHEAD + kc * 32, DHEAD);

  float mstat[8], lstat[8];
  v8f accO[8];
#pragma unroll
  for (int v = 0; v < 8; ++v) { mstat[v] = -1e30f; lstat[v] = 0.f; }
#pragma unroll
  for (int t = 0; t < 8; ++t)
#pragma unroll
    for (int v = 0; v < 8; ++v) accO[t][v] = 0.f;

  const int nkb = 2 * qb + 2; // causal: keys 0 .. qb*128+127
  for (int kb = 0; kb < nkb; ++kb) {
    const int key0 = kb * KBLK;
#if HAVE_TDM
    // K tile via TDM: 64 keys x 128 halfs, stride DH; pad 64 DW -> +4 DW (LKD=136)
    if (wave == 0)
      tdm_load_2d_f16(ldsK, Kh + (long)key0 * DHEAD,
                      /*tensor_d0=*/DHEAD, /*tensor_d1=*/KBLK, /*stride0=*/DHEAD,
                      /*tile_d0=*/DHEAD, /*tile_d1=*/KBLK,
                      /*pad_en=*/1u, /*pad_int=*/5u /*64 DW*/, /*pad_amt=*/3u /*4 DW*/);
#else
    { // K tile: [key][dh], 64 x 128; 32 halfs per thread
      const int kk = tid >> 2;
      const int p  = (tid & 3) * 32;
      const half_t* gp = Kh + (long)(key0 + kk) * DHEAD + p;
#pragma unroll
      for (int c = 0; c < 4; ++c)
        *(v8h*)(sK + kk * LKD + p + c * 8) = *(const v8h*)(gp + c * 8);
    }
#endif
    { // V tile transposed: read V[key][dh] coalesced, write sVt[dh][key]
      const int kk = tid >> 2;
      const int p  = (tid & 3) * 32;
      const half_t* gp = Vh + (long)(key0 + kk) * DHEAD + p;
#pragma unroll
      for (int c = 0; c < 4; ++c) {
        v8h x = *(const v8h*)(gp + c * 8);
#pragma unroll
        for (int i = 0; i < 8; ++i)
          sVt[(p + c * 8 + i) * LVD + kk] = x[i];
      }
    }
#if HAVE_TDM
    if (wave == 0) __builtin_amdgcn_s_wait_tensorcnt(0);
#endif
    __syncthreads();

    // S = Q * K^T : 16 rows x 64 keys (4 col tiles), K-dim = 128 (4 chunks)
    v8f sc[4];
#pragma unroll
    for (int ct = 0; ct < 4; ++ct) {
      v8f c;
#pragma unroll
      for (int v = 0; v < 8; ++v) c[v] = 0.f;
#pragma unroll
      for (int kc = 0; kc < 4; ++kc) {
        v16h bf = frag_b(sK + (ct * 16) * LKD + kc * 32, LKD);
        c = __builtin_amdgcn_wmma_f32_16x16x32_f16(false, qf[kc], false, bf,
                                                   (short)0, c, false, false);
      }
      sc[ct] = c;
    }

    // scale + causal mask + running row max
    float nmax[8];
#pragma unroll
    for (int v = 0; v < 8; ++v) nmax[v] = mstat[v];
#pragma unroll
    for (int ct = 0; ct < 4; ++ct) {
      const int keyg = key0 + ct * 16 + cn;
#pragma unroll
      for (int v = 0; v < 8; ++v) {
        const int qg = qrow0 + g * 8 + v;
        float x = sc[ct][v] * SOFT_SCALE;
        if (keyg > qg) x = -1e30f;
        sc[ct][v] = x;
        nmax[v] = fmaxf(nmax[v], x);
      }
    }
#pragma unroll
    for (int v = 0; v < 8; ++v) {
      float mv = nmax[v];
      mv = fmaxf(mv, __shfl_xor(mv, 1, 32));
      mv = fmaxf(mv, __shfl_xor(mv, 2, 32));
      mv = fmaxf(mv, __shfl_xor(mv, 4, 32));
      mv = fmaxf(mv, __shfl_xor(mv, 8, 32));
      nmax[v] = mv;
    }
    float alpha[8], rsum[8];
#pragma unroll
    for (int v = 0; v < 8; ++v) {
      alpha[v] = __expf(mstat[v] - nmax[v]);
      mstat[v] = nmax[v];
      rsum[v]  = 0.f;
    }
    // P = exp(S - m); write C-layout -> A-layout via per-wave LDS region
#pragma unroll
    for (int ct = 0; ct < 4; ++ct) {
#pragma unroll
      for (int v = 0; v < 8; ++v) {
        const float pval = __expf(sc[ct][v] - mstat[v]);
        rsum[v] += pval;
        sP[(wave * 16 + g * 8 + v) * LPD + ct * 16 + cn] = (half_t)pval;
      }
    }
#pragma unroll
    for (int v = 0; v < 8; ++v) {
      float rv = rsum[v];
      rv += __shfl_xor(rv, 1, 32);
      rv += __shfl_xor(rv, 2, 32);
      rv += __shfl_xor(rv, 4, 32);
      rv += __shfl_xor(rv, 8, 32);
      lstat[v] = lstat[v] * alpha[v] + rv;
    }
#pragma unroll
    for (int t = 0; t < 8; ++t)
#pragma unroll
      for (int v = 0; v < 8; ++v) accO[t][v] *= alpha[v];

    // wave-local LDS produce->consume: drain DS counter (no full barrier needed)
    asm volatile("s_wait_dscnt 0" ::: "memory");

    // O += P * V : K-dim = 64 keys (2 chunks), 8 dh col tiles
    v16h pf[2];
#pragma unroll
    for (int kc = 0; kc < 2; ++kc)
      pf[kc] = frag_a(sP + wave * 16 * LPD + kc * 32, LPD);
#pragma unroll
    for (int dt = 0; dt < 8; ++dt) {
#pragma unroll
      for (int kc = 0; kc < 2; ++kc) {
        v16h bf = frag_b(sVt + (dt * 16) * LVD + kc * 32, LVD);
        accO[dt] = __builtin_amdgcn_wmma_f32_16x16x32_f16(false, pf[kc], false, bf,
                                                          (short)0, accO[dt], false, false);
      }
    }
    __syncthreads(); // before next tile overwrites sK / sVt
  }

  // normalize and store concat-heads layout [S][H*DH]
#pragma unroll
  for (int dt = 0; dt < 8; ++dt) {
#pragma unroll
    for (int v = 0; v < 8; ++v) {
      const int r = qrow0 + g * 8 + v;
      const int c = h * DHEAD + dt * 16 + cn;
      const float o = accO[dt][v] / lstat[v];
      attnOut[(long)r * EMB + c] = (half_t)o;
    }
  }
}

// ---------------------------------------------------------------------------
// Host-side launch
// ---------------------------------------------------------------------------
extern "C" void kernel_launch(void* const* d_in, const int* in_sizes, int n_in,
                              void* d_out, int out_size, void* d_ws, size_t ws_size,
                              hipStream_t stream)
{
  (void)in_sizes; (void)n_in; (void)out_size; (void)ws_size;

  const float* x  = (const float*)d_in[0];
  const float* Wq = (const float*)d_in[1];
  const float* bq = (const float*)d_in[2];
  const float* Wk = (const float*)d_in[3];
  const float* bk = (const float*)d_in[4];
  const float* Wv = (const float*)d_in[5];
  const float* bv = (const float*)d_in[6];
  const float* WO = (const float*)d_in[7];
  const float* bo = (const float*)d_in[8];
  float* out = (float*)d_out;

  char* ws = (char*)d_ws;
  const size_t SEG = (size_t)S_LEN * EMB * sizeof(half_t); // 8 MiB per buffer
  half_t* xh   = (half_t*)(ws + 0 * SEG);
  half_t* wqh  = (half_t*)(ws + 1 * SEG);
  half_t* wkh  = (half_t*)(ws + 2 * SEG);
  half_t* wvh  = (half_t*)(ws + 3 * SEG);
  half_t* woh  = (half_t*)(ws + 4 * SEG);
  half_t* qbuf = (half_t*)(ws + 5 * SEG);
  half_t* kbuf = (half_t*)(ws + 6 * SEG);
  half_t* vbuf = (half_t*)(ws + 7 * SEG);
  half_t* attn = (half_t*)(ws + 8 * SEG);

  const int NELEM = S_LEN * EMB; // 4M (W tensors are also 16*2048*128 = 4M)
  cast_f32_to_f16<<<2048, 256, 0, stream>>>(x,  xh,  NELEM);
  cast_f32_to_f16<<<2048, 256, 0, stream>>>(Wq, wqh, NELEM);
  cast_f32_to_f16<<<2048, 256, 0, stream>>>(Wk, wkh, NELEM);
  cast_f32_to_f16<<<2048, 256, 0, stream>>>(Wv, wvh, NELEM);
  cast_f32_to_f16<<<2048, 256, 0, stream>>>(WO, woh, NELEM);

  const long wstride = (long)EMB * DHEAD;    // per-head weight stride
  const long ostride = (long)S_LEN * DHEAD;  // per-head output stride
  dim3 gq(S_LEN / 128, 1, NH);
  gemm_f16_wmma<<<gq, 256, 0, stream>>>(xh, EMB, wqh, DHEAD, wstride, bq, DHEAD,
                                        qbuf, nullptr, DHEAD, ostride, EMB);
  gemm_f16_wmma<<<gq, 256, 0, stream>>>(xh, EMB, wkh, DHEAD, wstride, bk, DHEAD,
                                        kbuf, nullptr, DHEAD, ostride, EMB);
  gemm_f16_wmma<<<gq, 256, 0, stream>>>(xh, EMB, wvh, DHEAD, wstride, bv, DHEAD,
                                        vbuf, nullptr, DHEAD, ostride, EMB);

  dim3 ga(S_LEN / 128, NH);
  flash_attn_f16<<<ga, 256, 0, stream>>>(qbuf, kbuf, vbuf, attn);

  dim3 go(S_LEN / 128, EMB / 128, 1);
  gemm_f16_wmma<<<go, 256, 0, stream>>>(attn, EMB, woh, EMB, 0, bo, 0,
                                        nullptr, out, EMB, 0, EMB);
}
